// SpikeRNN2D_317827580119
// MI455X (gfx1250) — compile-verified, matrix operand order checked by reading, not run on hardware
//
#include <hip/hip_runtime.h>

typedef __attribute__((ext_vector_type(16))) _Float16 v16h;
typedef __attribute__((ext_vector_type(8)))  float    v8f;

#define BS   64
#define LEN  128
#define CN   32
#define HID  256
#define NROWS 2048            // BS*CN
#define ROWS_PER_BLOCK 16
#define TCHUNK 16             // 256 threads = 16 rows x 16 lookahead timesteps
#define APITCH 264            // 256 + 8 halves pad -> 528B pitch, 4-bank stride
#define MTILES 2              // m-tiles per wave: m = wave, wave+8

// One RNN timestep: load A fragments, dual interleaved WMMA chains, LIF.
template<bool STORE>
__device__ __forceinline__ void gemm_step(
    const _Float16* __restrict__ abase,
    const v16h (&bfrag)[MTILES][8],
    v8f (&vstate)[MTILES], const v8f (&biasv)[MTILES],
    int wave, int lm, int hi, int lane,
    int nbase, float* __restrict__ out)
{
    // A fragments: 16 spike rows x K=256, 8 chunks of K=32 (shared by all waves)
    v16h afrag[8];
    #pragma unroll
    for (int kc = 0; kc < 8; ++kc) {
        union { uint4 q[2]; v16h v; } u;
        u.q[0] = *(const uint4*)(abase + kc * 32 + hi);
        u.q[1] = *(const uint4*)(abase + kc * 32 + hi + 16);
        afrag[kc] = u.v;
    }
    // Two independent accumulator chains, interleaved to hide XDL latency.
    v8f acc0 = biasv[0];
    v8f acc1 = biasv[1];
    #pragma unroll
    for (int kc = 0; kc < 8; ++kc) {
        acc0 = __builtin_amdgcn_wmma_f32_16x16x32_f16(
            false, afrag[kc], false, bfrag[0][kc], (short)0, acc0, false, false);
        acc1 = __builtin_amdgcn_wmma_f32_16x16x32_f16(
            false, afrag[kc], false, bfrag[1][kc], (short)0, acc1, false, false);
    }
    // Stateful LIF on both 16x16 tiles (D layout: vgpr i -> row, lane -> col)
    #pragma unroll
    for (int mi = 0; mi < MTILES; ++mi) {
        const v8f& acc = (mi == 0) ? acc0 : acc1;
        v8f vs = vstate[mi];
        #pragma unroll
        for (int i = 0; i < 8; ++i) {
            float vv = 0.5f * (vs[i] + acc[i]);
            bool fire = (vv >= 1.0f);
            vs[i] = fire ? 0.0f : vv;
            if (STORE) {
                int orow = nbase + i + ((lane >= 16) ? 8 : 0);
                int ocol = (wave + mi * 8) * 16 + lm;
                int idx  = orow * HID + ocol;
                float s  = fire ? 1.0f : 0.0f;
                out[idx] = s;                 // output 0
                out[NROWS * HID + idx] = s;   // output 1 (same data)
            }
        }
        vstate[mi] = vs;
    }
}

template<bool LAST>
__device__ __forceinline__ void rnn_chunk(
    const _Float16* __restrict__ a_stage,
    const v16h (&bfrag)[MTILES][8],
    v8f (&vstate)[MTILES], const v8f (&biasv)[MTILES],
    int wave, int lm, int hi, int lane,
    int nbase, float* __restrict__ out)
{
    const int nloop = LAST ? (TCHUNK - 1) : TCHUNK;
    #pragma unroll 1
    for (int dt = 0; dt < nloop; ++dt) {
        gemm_step<false>(a_stage + (dt * ROWS_PER_BLOCK + lm) * APITCH,
                         bfrag, vstate, biasv, wave, lm, hi, lane, nbase, out);
    }
    if (LAST) {  // peeled final timestep: stores enabled, no per-element checks
        gemm_step<true>(a_stage + ((TCHUNK - 1) * ROWS_PER_BLOCK + lm) * APITCH,
                        bfrag, vstate, biasv, wave, lm, hi, lane, nbase, out);
    }
}

__launch_bounds__(256, 1)
__global__ void snn_fused(const float* __restrict__ x,
                          const float* __restrict__ conv_w,
                          const float* __restrict__ conv_b,
                          const float* __restrict__ bn_g,
                          const float* __restrict__ bn_b,
                          const float* __restrict__ bn_m,
                          const float* __restrict__ bn_v,
                          const float* __restrict__ lin_w,
                          const float* __restrict__ lin_b,
                          float* __restrict__ out)
{
    // 132KB spike staging + 4KB packed coefs = 136KB LDS (2 blocks/WGP)
    __shared__ alignas(16) _Float16 a_stage[TCHUNK * ROWS_PER_BLOCK * APITCH];
    __shared__ alignas(16) float4  coef[HID];

    const int tid   = threadIdx.x;
    const int nbase = blockIdx.x * ROWS_PER_BLOCK;

    // ---- fold conv + BN into one float4 per hidden channel ----
    {
        int h = tid; // 256 threads == HID
        float inv = bn_g[h] * rsqrtf(bn_v[h] + 1e-5f);
        float4 k;
        k.x = inv * conv_w[h * 3 + 0];
        k.y = inv * conv_w[h * 3 + 1];
        k.z = inv * conv_w[h * 3 + 2];
        k.w = inv * (conv_b[h] - bn_m[h]) + bn_b[h];
        coef[h] = k;
    }

    // ---- GEMM wave mapping: each wave owns m-tiles {wave, wave+8} ----
    const int lane = tid & 31;
    const int wave = tid >> 5;
    const int lm   = lane & 15;
    const int hi   = (lane >= 16) ? 8 : 0;  // K sub-offset within 32-K chunk

    // ---- B fragments resident in registers: load lin_w f32, convert to fp16.
    //      B = lin_w^T, so column n of B is row n of lin_w (K contiguous).
    v16h bfrag[MTILES][8];
    #pragma unroll
    for (int mi = 0; mi < MTILES; ++mi) {
        const int m = wave + mi * 8;
        const float* wr0 = lin_w + (m * 16 + lm) * HID;
        #pragma unroll
        for (int kc = 0; kc < 8; ++kc) {
            const float* wr = wr0 + kc * 32 + hi;
            union { v16h v; _Float16 e[16]; } ub;
            #pragma unroll
            for (int j = 0; j < 8; ++j) ub.e[j]     = (_Float16)wr[j];
            #pragma unroll
            for (int j = 0; j < 8; ++j) ub.e[8 + j] = (_Float16)wr[16 + j];
            bfrag[mi][kc] = ub.v;
        }
    }

    // bias vectors (broadcast across the 8 D rows; column depends on lane only)
    v8f biasv[MTILES];
    #pragma unroll
    for (int mi = 0; mi < MTILES; ++mi) {
        float bv = lin_b[(wave + mi * 8) * 16 + lm];
        #pragma unroll
        for (int i = 0; i < 8; ++i) biasv[mi][i] = bv;
    }

    // ---- encoder mapping: 256 threads = 16 rows x 16 lookahead timesteps ----
    const int erow = tid & 15;
    const int edt  = tid >> 4;
    const int n = nbase + erow;
    const int b = n >> 5;            // n = b*32 + c
    const int c = n & 31;

    v8f vstate[MTILES];
    #pragma unroll
    for (int mi = 0; mi < MTILES; ++mi) vstate[mi] = {};

    __syncthreads();  // coef ready

    for (int t0 = 0; t0 < LEN; t0 += TCHUNK) {
        // ---- encoder: LIF scan over hidden for (row erow, time t0+edt) ----
        {
            const int tt = t0 + edt;
            float xm1 = (tt > 0)       ? x[(b * LEN + tt - 1) * CN + c] : 0.0f;
            float x0  =                  x[(b * LEN + tt    ) * CN + c];
            float xp1 = (tt < LEN - 1) ? x[(b * LEN + tt + 1) * CN + c] : 0.0f;
            _Float16* arow = &a_stage[(edt * ROWS_PER_BLOCK + erow) * APITCH];
            float v = 0.0f;
            for (int h8 = 0; h8 < HID; h8 += 8) {
                union { uint4 q; _Float16 e[8]; } pk;
                #pragma unroll
                for (int j = 0; j < 8; ++j) {
                    float4 k = coef[h8 + j];
                    float enc = k.x * xm1 + k.y * x0 + k.z * xp1 + k.w;
                    v = 0.5f * (v + enc);                // v += (x - v)/tau
                    bool fire = (v >= 1.0f);
                    pk.e[j] = fire ? (_Float16)1.0f : (_Float16)0.0f;
                    v = fire ? 0.0f : v;                 // hard reset
                }
                *(uint4*)(arow + h8) = pk.q;             // ds_store_b128
            }
        }
        __syncthreads();  // spikes visible to GEMM waves

        // ---- GEMM + LIF for TCHUNK timesteps ----
        if (t0 + TCHUNK < LEN) {
            rnn_chunk<false>(a_stage, bfrag, vstate, biasv,
                             wave, lm, hi, lane, nbase, out);
        } else {
            rnn_chunk<true>(a_stage, bfrag, vstate, biasv,
                            wave, lm, hi, lane, nbase, out);
        }
        __syncthreads();  // GEMM done reading a_stage before next encoder pass
    }
}

extern "C" void kernel_launch(void* const* d_in, const int* in_sizes, int n_in,
                              void* d_out, int out_size, void* d_ws, size_t ws_size,
                              hipStream_t stream) {
    const float* x      = (const float*)d_in[0];
    const float* conv_w = (const float*)d_in[1];
    const float* conv_b = (const float*)d_in[2];
    const float* bn_g   = (const float*)d_in[3];
    const float* bn_b   = (const float*)d_in[4];
    const float* bn_m   = (const float*)d_in[5];
    const float* bn_v   = (const float*)d_in[6];
    const float* lin_w  = (const float*)d_in[7];
    const float* lin_b  = (const float*)d_in[8];
    snn_fused<<<NROWS / ROWS_PER_BLOCK, 256, 0, stream>>>(
        x, conv_w, conv_b, bn_g, bn_b, bn_m, bn_v, lin_w, lin_b, (float*)d_out);
}
